// CausalVarlenSelfAttention_15152644620516
// MI455X (gfx1250) — matrix operand order repaired and checked by reading, not compiled, not run on hardware
//
#include <hip/hip_runtime.h>
#include <hip/hip_bf16.h>

#define DEVI __device__ __forceinline__

typedef unsigned short u16;
typedef __attribute__((ext_vector_type(16))) __bf16 v16bf;
typedef __attribute__((ext_vector_type(8)))  float  v8f;

union FragBF { v16bf v; uint4 q[2]; u16 s[16]; };

DEVI u16 f2bf_bits(float f) {
    unsigned int u = __builtin_bit_cast(unsigned int, f);
    unsigned int r = u + 0x7FFFu + ((u >> 16) & 1u);
    return (u16)(r >> 16);
}

DEVI v8f zero8() {
    v8f z;
#pragma unroll
    for (int i = 0; i < 8; ++i) z[i] = 0.f;
    return z;
}

// A fragment: 16(M) x 32(K) bf16.  lane m = m0 + (lane&15); k chunks
// {kb..kb+7, kb+16..kb+23} with kb = k0 + (lane>=16 ? 8 : 0).
DEVI v16bf load_frag_a(const u16* base, int ld, int m0, int k0, int lane) {
    int m  = m0 + (lane & 15);
    int kb = k0 + ((lane & 16) ? 8 : 0);
    const u16* p = base + (size_t)m * ld + kb;
    FragBF f;
    f.q[0] = *reinterpret_cast<const uint4*>(p);
    f.q[1] = *reinterpret_cast<const uint4*>(p + 16);
    return f.v;
}

// B fragment: 32(K) x 16(N) bf16, stored as rows of N with K contiguous
// (i.e. B^T row-major).  lane n = n0 + (lane&15); k chunks
// {kb..kb+7, kb+16..kb+23} with kb = k0 + (lane>=16 ? 8 : 0).
DEVI v16bf load_frag_b(const u16* base, int ld, int n0, int k0, int lane) {
    int n  = n0 + (lane & 15);
    int kb = k0 + ((lane & 16) ? 8 : 0);
    const u16* p = base + (size_t)n * ld + kb;
    FragBF f;
    f.q[0] = *reinterpret_cast<const uint4*>(p);
    f.q[1] = *reinterpret_cast<const uint4*>(p + 16);
    return f.v;
}

DEVI v8f wmma_bf16(v16bf a, v16bf b, v8f c) {
    return __builtin_amdgcn_wmma_f32_16x16x32_bf16(
        false, a, false, b, (short)0, c, false, false);
}

// ---------------------------------------------------------------- convert
__global__ void cvt_f32_bf16_kernel(const float* __restrict__ src,
                                    u16* __restrict__ dst, int n4) {
    int i = blockIdx.x * blockDim.x + threadIdx.x;
    int stride = gridDim.x * blockDim.x;
    for (; i < n4; i += stride) {
        float4 v = reinterpret_cast<const float4*>(src)[i];
        unsigned long long pk =
              (unsigned long long)f2bf_bits(v.x)
            | ((unsigned long long)f2bf_bits(v.y) << 16)
            | ((unsigned long long)f2bf_bits(v.z) << 32)
            | ((unsigned long long)f2bf_bits(v.w) << 48);
        reinterpret_cast<unsigned long long*>(dst)[i] = pk;
    }
}

// ------------------------------------------------- QKV GEMM + RoPE + RMSNorm
// grid (32 m-blocks, 48 head-slots), block 128 (4 waves).
// Each wave computes a 32(M) x 128(N) tile.  All 8 B fragments are loaded
// into distinct registers first so the 16 WMMAs overlap the load latency.
__global__ __launch_bounds__(128)
void qkv_kernel(const u16* __restrict__ xb, const u16* __restrict__ wb,
                const float* __restrict__ cosc, const float* __restrict__ sinc,
                const float* __restrict__ norm_w,
                u16* __restrict__ qb, u16* __restrict__ kbuf,
                u16* __restrict__ vt) {
    const int lane = threadIdx.x & 31;
    const int w    = threadIdx.x >> 5;
    const int m0   = blockIdx.x * 128 + w * 32;
    const int slot = blockIdx.y;

    v8f acc[2][8];
#pragma unroll
    for (int mt = 0; mt < 2; ++mt)
#pragma unroll
        for (int t = 0; t < 8; ++t) acc[mt][t] = zero8();

    const u16* wslot = wb + (size_t)slot * 128 * 2048;
    const u16* pfa0 = xb + (size_t)(m0      + (lane & 15)) * 2048;
    const u16* pfa1 = xb + (size_t)(m0 + 16 + (lane & 15)) * 2048;

    for (int k0 = 0; k0 < 2048; k0 += 32) {
        int kpf = (k0 + 64) & 2047;   // wrapped: always valid, no branch
        __builtin_prefetch(pfa0 + kpf, 0, 3);
        __builtin_prefetch(pfa1 + kpf, 0, 3);

        v16bf a0 = load_frag_a(xb, 2048, m0,      k0, lane);
        v16bf a1 = load_frag_a(xb, 2048, m0 + 16, k0, lane);
        v16bf bfr[8];
#pragma unroll
        for (int t = 0; t < 8; ++t)
            bfr[t] = load_frag_b(wslot, 2048, t * 16, k0, lane);
#pragma unroll
        for (int t = 0; t < 8; ++t) {
            acc[0][t] = wmma_bf16(a0, bfr[t], acc[0][t]);
            acc[1][t] = wmma_bf16(a1, bfr[t], acc[1][t]);
        }
    }

    const int hi  = (lane >> 4) & 1;
    const int nlo = lane & 15;

#pragma unroll
    for (int mt = 0; mt < 2; ++mt) {
        const int mb = m0 + mt * 16;
        if (slot < 32) {  // q or k: RoPE then RMSNorm
#pragma unroll
            for (int r = 0; r < 8; ++r) {
                int tok = mb + r + hi * 8;
                int pos = tok & 1023;
#pragma unroll
                for (int t = 0; t < 4; ++t) {
                    int d   = t * 16 + nlo;          // d in [0,64)
                    float c = cosc[pos * 64 + d];
                    float s = sinc[pos * 64 + d];
                    float x1 = acc[mt][t][r], x2 = acc[mt][t + 4][r];
                    acc[mt][t][r]     = x1 * c + x2 * s;
                    acc[mt][t + 4][r] = x2 * c - x1 * s;
                }
                float ss = 0.f;
#pragma unroll
                for (int t = 0; t < 8; ++t) ss += acc[mt][t][r] * acc[mt][t][r];
                ss += __shfl_xor(ss, 1, 32);
                ss += __shfl_xor(ss, 2, 32);
                ss += __shfl_xor(ss, 4, 32);
                ss += __shfl_xor(ss, 8, 32);
                float inv = rsqrtf(ss * (1.f / 128.f) + 1e-6f);
#pragma unroll
                for (int t = 0; t < 8; ++t)
                    acc[mt][t][r] *= inv * norm_w[t * 16 + nlo];
            }
            int h = slot & 15;
            u16* dst = (slot < 16) ? qb : kbuf;
#pragma unroll
            for (int r = 0; r < 8; ++r) {
                int tok = mb + r + hi * 8;
                int b = tok >> 10, s = tok & 1023;
                size_t base = ((size_t)(b * 16 + h) * 1024 + s) * 128;
#pragma unroll
                for (int t = 0; t < 8; ++t)
                    dst[base + t * 16 + nlo] = f2bf_bits(acc[mt][t][r]);
            }
        } else {  // v: store transposed [b,h,d,s]
            int h = slot - 32;
#pragma unroll
            for (int r = 0; r < 8; ++r) {
                int tok = mb + r + hi * 8;
                int b = tok >> 10, s = tok & 1023;
                size_t hb = (size_t)(b * 16 + h) * 128;
#pragma unroll
                for (int t = 0; t < 8; ++t) {
                    int d = t * 16 + nlo;
                    vt[(hb + d) * 1024 + s] = f2bf_bits(acc[mt][t][r]);
                }
            }
        }
    }
}

// ---------------------------------------------------------------- gate
// grid 4096 (tokens), block 512 = 16 waves, wave h computes one head dot.
__global__ __launch_bounds__(512)
void gate_kernel(const float* __restrict__ x, const float* __restrict__ gw,
                 const float* __restrict__ gb, float* __restrict__ gate) {
    int tok  = blockIdx.x;
    int h    = threadIdx.x >> 5;
    int lane = threadIdx.x & 31;
    const float* xr = x + (size_t)tok * 2048;
    const float* wr = gw + (size_t)h * 2048;
    float dot = 0.f;
    for (int j = lane; j < 2048; j += 32) dot += xr[j] * wr[j];
    dot += __shfl_xor(dot, 1, 32);
    dot += __shfl_xor(dot, 2, 32);
    dot += __shfl_xor(dot, 4, 32);
    dot += __shfl_xor(dot, 8, 32);
    dot += __shfl_xor(dot, 16, 32);
    if (lane == 0)
        gate[tok * 16 + h] = 1.f / (1.f + __expf(-(dot + gb[h])));
}

// --------------------------------------------------- flash attention + gate
// grid (16 q-blocks, 64 head-batches), block 128 (4 waves); wave = 16 q rows.
__global__ __launch_bounds__(128)
void attn_kernel(const u16* __restrict__ qb, const u16* __restrict__ kbuf,
                 const u16* __restrict__ vt, const float* __restrict__ gate,
                 u16* __restrict__ ag) {
    __shared__ float pbuf[4][16][33];

    const int lane = threadIdx.x & 31;
    const int w    = threadIdx.x >> 5;
    const int bh   = blockIdx.y;            // 0..63
    const int b    = bh >> 4, h = bh & 15;
    const int q0   = blockIdx.x * 64 + w * 16;
    const int hi   = (lane >> 4) & 1;
    const int nlo  = lane & 15;

    const u16* qh = qb   + (size_t)bh * 1024 * 128;
    const u16* kh = kbuf + (size_t)bh * 1024 * 128;
    const u16* vh = vt   + (size_t)bh * 128 * 1024;

    v16bf aq[4];
#pragma unroll
    for (int kk = 0; kk < 4; ++kk) aq[kk] = load_frag_a(qh, 128, q0, kk * 32, lane);

    v8f acc[8];
#pragma unroll
    for (int t = 0; t < 8; ++t) acc[t] = zero8();
    float mi[8], li[8];
#pragma unroll
    for (int r = 0; r < 8; ++r) { mi[r] = -1e30f; li[r] = 0.f; }

    const float scale = 0.08838834764831845f;  // 1/sqrt(128)
    const int nkb = (q0 + 16 + 31) >> 5;       // 32-key blocks (causal bound)

    for (int kb = 0; kb < nkb; ++kb) {
        const int kbase = kb * 32;
        {   // branch-free prefetch of next key block (clamped addresses)
            int pr = min(kbase + 32 + lane, 1023);
            int pc = min(kbase + 32, 992);
            __builtin_prefetch(kh + (size_t)pr * 128, 0, 3);
            __builtin_prefetch(vh + (size_t)(lane * 4) * 1024 + pc, 0, 3);
        }
        // load all 8 K fragments, then 8 score WMMAs
        v16bf bk[8];
#pragma unroll
        for (int kk = 0; kk < 4; ++kk) {
            bk[kk * 2]     = load_frag_b(kh, 128, kbase,      kk * 32, lane);
            bk[kk * 2 + 1] = load_frag_b(kh, 128, kbase + 16, kk * 32, lane);
        }
        v8f s0 = zero8(), s1 = zero8();
#pragma unroll
        for (int kk = 0; kk < 4; ++kk) {
            s0 = wmma_bf16(aq[kk], bk[kk * 2],     s0);
            s1 = wmma_bf16(aq[kk], bk[kk * 2 + 1], s1);
        }
        // online softmax update
#pragma unroll
        for (int r = 0; r < 8; ++r) {
            int qg  = q0 + r + hi * 8;
            int k0g = kbase + nlo, k1g = kbase + 16 + nlo;
            float v0 = (k0g <= qg) ? s0[r] * scale : -1e30f;
            float v1 = (k1g <= qg) ? s1[r] * scale : -1e30f;
            float mrow = fmaxf(v0, v1);
            mrow = fmaxf(mrow, __shfl_xor(mrow, 1, 32));
            mrow = fmaxf(mrow, __shfl_xor(mrow, 2, 32));
            mrow = fmaxf(mrow, __shfl_xor(mrow, 4, 32));
            mrow = fmaxf(mrow, __shfl_xor(mrow, 8, 32));
            float mnew = fmaxf(mi[r], mrow);
            float a    = __expf(mi[r] - mnew);
            float p0   = __expf(v0 - mnew);
            float p1   = __expf(v1 - mnew);
            float ps   = p0 + p1;
            ps += __shfl_xor(ps, 1, 32);
            ps += __shfl_xor(ps, 2, 32);
            ps += __shfl_xor(ps, 4, 32);
            ps += __shfl_xor(ps, 8, 32);
            li[r] = li[r] * a + ps;
            mi[r] = mnew;
#pragma unroll
            for (int t = 0; t < 8; ++t) acc[t][r] *= a;
            int M = r + hi * 8;
            pbuf[w][M][nlo]      = p0;
            pbuf[w][M][16 + nlo] = p1;
        }
        asm volatile("" ::: "memory");  // keep LDS store->load order (wave-private)
        // re-layout P (16x32 D tiles) into an A fragment via LDS
        FragBF pf;
        int akb = hi ? 8 : 0;
#pragma unroll
        for (int j = 0; j < 8; ++j) pf.s[j]     = f2bf_bits(pbuf[w][nlo][akb + j]);
#pragma unroll
        for (int j = 0; j < 8; ++j) pf.s[8 + j] = f2bf_bits(pbuf[w][nlo][akb + 16 + j]);
        // P @ V: load all 8 V fragments, then 8 WMMAs
        v16bf bv[8];
#pragma unroll
        for (int t = 0; t < 8; ++t)
            bv[t] = load_frag_b(vh, 1024, t * 16, kbase, lane);
#pragma unroll
        for (int t = 0; t < 8; ++t)
            acc[t] = wmma_bf16(pf.v, bv[t], acc[t]);
        asm volatile("" ::: "memory");
    }
    // epilogue: normalize, gate, store bf16
#pragma unroll
    for (int r = 0; r < 8; ++r) {
        int qg  = q0 + r + hi * 8;
        int tok = b * 1024 + qg;
        float g = gate[tok * 16 + h] / li[r];
        size_t base = (size_t)tok * 2048 + h * 128;
#pragma unroll
        for (int t = 0; t < 8; ++t)
            ag[base + t * 16 + nlo] = f2bf_bits(acc[t][r] * g);
    }
}

// ---------------------------------------------------------------- out proj
// grid (32 m-blocks, 16 n-blocks), block 128; wave = 32(M) x 128(N) f32 tile.
__global__ __launch_bounds__(128)
void oproj_kernel(const u16* __restrict__ agb, const u16* __restrict__ wob,
                  float* __restrict__ out) {
    const int lane = threadIdx.x & 31;
    const int w    = threadIdx.x >> 5;
    const int m0   = blockIdx.x * 128 + w * 32;
    const int n0   = blockIdx.y * 128;

    v8f acc[2][8];
#pragma unroll
    for (int mt = 0; mt < 2; ++mt)
#pragma unroll
        for (int t = 0; t < 8; ++t) acc[mt][t] = zero8();

    const u16* pfa0 = agb + (size_t)(m0      + (lane & 15)) * 2048;
    const u16* pfa1 = agb + (size_t)(m0 + 16 + (lane & 15)) * 2048;

    for (int k0 = 0; k0 < 2048; k0 += 32) {
        int kpf = (k0 + 64) & 2047;
        __builtin_prefetch(pfa0 + kpf, 0, 3);
        __builtin_prefetch(pfa1 + kpf, 0, 3);

        v16bf a0 = load_frag_a(agb, 2048, m0,      k0, lane);
        v16bf a1 = load_frag_a(agb, 2048, m0 + 16, k0, lane);
        v16bf bfr[8];
#pragma unroll
        for (int t = 0; t < 8; ++t)
            bfr[t] = load_frag_b(wob, 2048, n0 + t * 16, k0, lane);
#pragma unroll
        for (int t = 0; t < 8; ++t) {
            acc[0][t] = wmma_bf16(a0, bfr[t], acc[0][t]);
            acc[1][t] = wmma_bf16(a1, bfr[t], acc[1][t]);
        }
    }
    const int hi = (lane >> 4) & 1, nlo = lane & 15;
#pragma unroll
    for (int mt = 0; mt < 2; ++mt) {
#pragma unroll
        for (int r = 0; r < 8; ++r) {
            int tok = m0 + mt * 16 + r + hi * 8;
#pragma unroll
            for (int t = 0; t < 8; ++t)
                out[(size_t)tok * 2048 + n0 + t * 16 + nlo] = acc[mt][t][r];
        }
    }
}

extern "C" void kernel_launch(void* const* d_in, const int* in_sizes, int n_in,
                              void* d_out, int out_size, void* d_ws, size_t ws_size,
                              hipStream_t stream) {
    (void)in_sizes; (void)n_in; (void)out_size; (void)ws_size;
    const float* x      = (const float*)d_in[0];
    const float* Wqkv   = (const float*)d_in[1];
    const float* Wo     = (const float*)d_in[2];
    const float* gate_w = (const float*)d_in[3];
    const float* gate_b = (const float*)d_in[4];
    const float* norm_w = (const float*)d_in[5];
    const float* cosc   = (const float*)d_in[6];
    const float* sinc   = (const float*)d_in[7];

    char* p = (char*)d_ws;
    u16* xb    = (u16*)p;  p += (size_t)4096 * 2048 * 2;
    u16* wqkvb = (u16*)p;  p += (size_t)6144 * 2048 * 2;
    u16* wob   = (u16*)p;  p += (size_t)2048 * 2048 * 2;
    u16* qb    = (u16*)p;  p += (size_t)64 * 1024 * 128 * 2;
    u16* kbuf  = (u16*)p;  p += (size_t)64 * 1024 * 128 * 2;
    u16* vt    = (u16*)p;  p += (size_t)64 * 128 * 1024 * 2;
    float* gate = (float*)p; p += (size_t)4096 * 16 * 4;
    u16* ag    = (u16*)p;

    cvt_f32_bf16_kernel<<<2048, 256, 0, stream>>>(x,    xb,    4096 * 2048 / 4);
    cvt_f32_bf16_kernel<<<2048, 256, 0, stream>>>(Wqkv, wqkvb, 6144 * 2048 / 4);
    cvt_f32_bf16_kernel<<<2048, 256, 0, stream>>>(Wo,   wob,   2048 * 2048 / 4);

    qkv_kernel<<<dim3(32, 48), 128, 0, stream>>>(xb, wqkvb, cosc, sinc, norm_w,
                                                 qb, kbuf, vt);
    gate_kernel<<<4096, 512, 0, stream>>>(x, gate_w, gate_b, gate);
    attn_kernel<<<dim3(16, 64), 128, 0, stream>>>(qb, kbuf, vt, gate, ag);
    oproj_kernel<<<dim3(32, 16), 128, 0, stream>>>(ag, wob, (float*)d_out);
}
